// C_Net_77807627534400
// MI455X (gfx1250) — matrix operand order repaired and checked by reading, not compiled, not run on hardware
//
#include <hip/hip_runtime.h>
#include <hip/hip_bf16.h>
#include <math.h>

// Problem constants (from reference): B=2, C=256, H=W=48, NCH=12
#define B_    2
#define C_    256
#define NPIX  2304          // H*W
#define NCHN  12
#define NEGL  (-1e30f)

typedef __attribute__((ext_vector_type(16))) _Float16 v16h;
typedef __attribute__((ext_vector_type(8)))  _Float16 v8h;
typedef __attribute__((ext_vector_type(8)))  float    v8f;

// 16-lane (half-wave) sum: xor masks < 16 stay inside each half,
// matching the WMMA C/D layout (row m in lanes 0-15 or 16-31).
__device__ __forceinline__ float redsum16(float x) {
#pragma unroll
  for (int m = 1; m < 16; m <<= 1) x += __shfl_xor(x, m, 32);
  return x;
}

// ---------------------------------------------------------------- kernel 0
__global__ void init_kernel(float* __restrict__ out, int n) {
  int i = blockIdx.x * blockDim.x + threadIdx.x;
  if (i < n) out[i] = -1.0f;
}

// ---------------------------------------------------------------- kernel 1
// Per (b,k): masks -> LDS, counts, per-channel masked means.
__global__ __launch_bounds__(256)
void stats_kernel(const float* __restrict__ gf, const float* __restrict__ rf,
                  const float* __restrict__ glab, const float* __restrict__ rlab,
                  float* __restrict__ cntg, float* __restrict__ cntr,
                  float* __restrict__ mg, float* __restrict__ mr) {
  __shared__ float sgm[NPIX];
  __shared__ float srm[NPIX];
  __shared__ float scnt[2];
  const int b = blockIdx.y, k = blockIdx.x + 1;
  const int tid = threadIdx.x;
  const size_t bk = (size_t)b * NCHN + k;
  if (tid == 0) { scnt[0] = 0.f; scnt[1] = 0.f; }
  __syncthreads();
  float lg = 0.f, lr = 0.f;
  for (int n = tid; n < NPIX; n += 256) {
    float g = glab[bk * NPIX + n] > 0.5f ? 1.f : 0.f;
    float r = rlab[bk * NPIX + n] > 0.5f ? 1.f : 0.f;
    sgm[n] = g; srm[n] = r; lg += g; lr += r;
  }
  atomicAdd(&scnt[0], lg);
  atomicAdd(&scnt[1], lr);
  __syncthreads();
  const float cg = scnt[0], cr = scnt[1];
  if (tid == 0) { cntg[bk] = cg; cntr[bk] = cr; }
  // one channel per thread; mask streamed from LDS
  const int c = tid;
  const float* gp = gf + ((size_t)b * C_ + c) * NPIX;
  const float* rp = rf + ((size_t)b * C_ + c) * NPIX;
  float sg = 0.f, sr = 0.f;
  for (int n = 0; n < NPIX; ++n) { sg += gp[n] * sgm[n]; sr += rp[n] * srm[n]; }
  mg[bk * C_ + c] = sg / fmaxf(cg, 1.f);
  mr[bk * C_ + c] = sr / fmaxf(cr, 1.f);
}

// ---------------------------------------------------------------- kernel 2
// Per (b,k,pixel): (feat-mean)*mask, unit-normalize over C, f16-convert and
// store directly in WMMA fragment layouts (A-layout for gray, B-layout for rgb).
// Fragment array layout: [bk][tile16][kc(8)][lane(32)][e(16)] halfs.
__device__ __forceinline__ void normalize_store(
    const float* __restrict__ feat, const float* __restrict__ lab,
    const float* __restrict__ mean, _Float16* __restrict__ dst,
    int b, int k, int n, int tx, int ty, float (*red)[32], bool a_layout) {
  const size_t bk = (size_t)b * NCHN + k;
  const float mv = (lab[bk * NPIX + n] > 0.5f) ? 1.f : 0.f;
  float vals[32];
  float ss = 0.f;
#pragma unroll
  for (int cc = 0; cc < 32; ++cc) {
    const int c = ty * 32 + cc;                       // uniform across the wave
    float v = (feat[((size_t)b * C_ + c) * NPIX + n] - mean[bk * C_ + c]) * mv;
    vals[cc] = v; ss += v * v;
  }
  red[ty][tx] = ss;
  __syncthreads();
  float tot = 0.f;
#pragma unroll
  for (int g = 0; g < 8; ++g) tot += red[g][tx];
  __syncthreads();
  const float sc = tot > 0.f ? rsqrtf(tot) : 1.0f;
  const int t16 = n >> 4, r = n & 15, kc = ty;
  _Float16* base = dst + (((bk * 144 + t16) * 8 + kc) * 32) * 16;
  if (a_layout) {
    // A 16x32 map: k = e + (e>=8 ? 8 : 0) + 8*half  (ISA 7.12.2)
    v8h p;
#pragma unroll
    for (int e = 0; e < 8; ++e) p[e] = (_Float16)(vals[e] * sc);        // k 0..7  -> lane r,    e0..7
    *(v8h*)(base + r * 16) = p;
#pragma unroll
    for (int e = 0; e < 8; ++e) p[e] = (_Float16)(vals[8 + e] * sc);    // k 8..15 -> lane 16+r, e0..7
    *(v8h*)(base + (16 + r) * 16) = p;
#pragma unroll
    for (int e = 0; e < 8; ++e) p[e] = (_Float16)(vals[16 + e] * sc);   // k16..23 -> lane r,    e8..15
    *(v8h*)(base + r * 16 + 8) = p;
#pragma unroll
    for (int e = 0; e < 8; ++e) p[e] = (_Float16)(vals[24 + e] * sc);   // k24..31 -> lane 16+r, e8..15
    *(v8h*)(base + (16 + r) * 16 + 8) = p;
  } else {
    // B 32x16 map: k = e + (e>=8 ? 8 : 0) + 8*half (same as A)
    v8h q;
#pragma unroll
    for (int e = 0; e < 8; ++e) q[e] = (_Float16)(vals[e] * sc);
    *(v8h*)(base + r * 16) = q;
#pragma unroll
    for (int e = 0; e < 8; ++e) q[e] = (_Float16)(vals[8 + e] * sc);
    *(v8h*)(base + (16 + r) * 16) = q;
#pragma unroll
    for (int e = 0; e < 8; ++e) q[e] = (_Float16)(vals[16 + e] * sc);
    *(v8h*)(base + r * 16 + 8) = q;
#pragma unroll
    for (int e = 0; e < 8; ++e) q[e] = (_Float16)(vals[24 + e] * sc);
    *(v8h*)(base + (16 + r) * 16 + 8) = q;
  }
}

__global__ __launch_bounds__(256)
void normalize_kernel(const float* __restrict__ gf, const float* __restrict__ rf,
                      const float* __restrict__ glab, const float* __restrict__ rlab,
                      const float* __restrict__ mg, const float* __restrict__ mr,
                      _Float16* __restrict__ gnw, _Float16* __restrict__ rnw) {
  __shared__ float red[8][32];
  const int b = blockIdx.z, k = blockIdx.y + 1;
  const int tx = threadIdx.x & 31, ty = threadIdx.x >> 5;
  const int n = blockIdx.x * 32 + tx;
  normalize_store(gf, glab, mg, gnw, b, k, n, tx, ty, red, true);
  normalize_store(rf, rlab, mr, rnw, b, k, n, tx, ty, red, false);
}

// ---------------------------------------------------------------- kernel 3
// One block per (b, k, 16-row i-tile). 8 waves split 144 j-tiles (18 each).
// WMMA GEMM fused with fixed-shift softmax (logits are cosine similarities,
// |x|<=1+eps, so exp(x-1) can never overflow -> no online max, no per-tile
// cross-lane reductions). B tiles are double-buffered (ping-pong) so tile
// n+1's 16-load clause overlaps tile n's WMMAs + epilogue VALU.
struct Aux { float m, i0, i1, i2; };

__device__ __forceinline__ void load_tile(const _Float16* __restrict__ rnw_bk,
                                          const float* __restrict__ rlab_bk,
                                          const float* __restrict__ img_b,
                                          int jt, int L, int l16,
                                          v16h bf[8], Aux& ax) {
  const v16h* bp = (const v16h*)(rnw_bk + (size_t)jt * 8 * 512);
#pragma unroll
  for (int kc = 0; kc < 8; ++kc) bf[kc] = bp[kc * 32 + L];
  const int j = jt * 16 + l16;
  ax.m = rlab_bk[j] > 0.5f ? 1.f : 0.f;
  ax.i0 = img_b[j];
  ax.i1 = img_b[NPIX + j];
  ax.i2 = img_b[2 * NPIX + j];
  if (jt + 16 < 144)   // warm L2/L0 two stripe-steps ahead -> global_prefetch_b8
    __builtin_prefetch((const void*)(rnw_bk + (size_t)(jt + 16) * 8 * 512 + (size_t)L * 16), 0, 3);
}

__device__ __forceinline__ void consume_tile(const v16h a[8], const v16h bf[8],
                                             const Aux& ax, float rsm[8],
                                             float ro0[8], float ro1[8], float ro2[8]) {
  v8f acc = {};
#pragma unroll
  for (int kc = 0; kc < 8; ++kc)
    acc = __builtin_amdgcn_wmma_f32_16x16x32_f16(
        /*neg_a=*/false, a[kc], /*neg_b=*/false, bf[kc],
        /*c_mod=*/(short)0, acc, /*reuse_a=*/false, /*reuse_b=*/false);
#pragma unroll
  for (int v = 0; v < 8; ++v) {           // row m = v + 8*half
    const float e = __expf(acc[v] - 1.0f) * ax.m;   // ax.m==0 kills masked-out j
    rsm[v] += e;
    ro0[v] += e * ax.i0;
    ro1[v] += e * ax.i1;
    ro2[v] += e * ax.i2;
  }
}

__global__ __launch_bounds__(256)
void corr_softmax_kernel(const float* __restrict__ img,
                         const float* __restrict__ glab,
                         const float* __restrict__ rlab,
                         const float* __restrict__ cntg,
                         const float* __restrict__ cntr,
                         const _Float16* __restrict__ gnw,
                         const _Float16* __restrict__ rnw,
                         float* __restrict__ out) {
  const int b = blockIdx.z;
  const int k = blockIdx.y + 1;
  const int it = blockIdx.x;
  const size_t bk = (size_t)b * NCHN + k;
  const float cg = cntg[bk], cr = cntr[bk];
  if (!(cg > 1.0f && cr > 1.0f)) return;    // invalid class: canvas stays -1

  const int tid = threadIdx.x;
  const int w = tid >> 5;        // wave id 0..7
  const int L = tid & 31;        // lane
  const int half = L >> 4;
  const int l16 = L & 15;

  const _Float16* rnw_bk = rnw + bk * (size_t)NPIX * C_;
  const float* rlab_bk = rlab + bk * NPIX;
  const float* img_b = img + (size_t)b * 3 * NPIX;

  // A fragments (16 rows of gray unit-features x K=256), resident in VGPRs.
  const v16h* ap = (const v16h*)(gnw + ((bk * 144 + it) * 8) * 512);
  v16h a[8];
#pragma unroll
  for (int kc = 0; kc < 8; ++kc) a[kc] = ap[kc * 32 + L];

  float rsm[8], ro0[8], ro1[8], ro2[8];
#pragma unroll
  for (int v = 0; v < 8; ++v) { rsm[v] = 0.f; ro0[v] = 0.f; ro1[v] = 0.f; ro2[v] = 0.f; }

  v16h bA[8], bB[8];
  Aux xA, xB;
  load_tile(rnw_bk, rlab_bk, img_b, /*jt=*/w, L, l16, bA, xA);
  for (int jj = 0; jj < 18; jj += 2) {
    load_tile(rnw_bk, rlab_bk, img_b, (jj + 1) * 8 + w, L, l16, bB, xB);
    consume_tile(a, bA, xA, rsm, ro0, ro1, ro2);
    if (jj + 2 < 18)
      load_tile(rnw_bk, rlab_bk, img_b, (jj + 2) * 8 + w, L, l16, bA, xA);
    consume_tile(a, bB, xB, rsm, ro0, ro1, ro2);
  }

  // one cross-lane reduction per block (not per tile)
#pragma unroll
  for (int v = 0; v < 8; ++v) {
    rsm[v] = redsum16(rsm[v]);
    ro0[v] = redsum16(ro0[v]);
    ro1[v] = redsum16(ro1[v]);
    ro2[v] = redsum16(ro2[v]);
  }

  // merge 8 waves' partial sums (fixed shift -> plain addition)
  __shared__ float sS[8][16], sO0[8][16], sO1[8][16], sO2[8][16];
  if (l16 == 0) {
#pragma unroll
    for (int v = 0; v < 8; ++v) {
      const int m = half * 8 + v;
      sS[w][m] = rsm[v];
      sO0[w][m] = ro0[v]; sO1[w][m] = ro1[v]; sO2[w][m] = ro2[v];
    }
  }
  __syncthreads();
  if (tid < 16) {
    const int i = it * 16 + tid;
    if (glab[bk * NPIX + i] > 0.5f) {        // write = gray-mask & valid
      float tot = 0.f, o0 = 0.f, o1 = 0.f, o2 = 0.f;
#pragma unroll
      for (int q = 0; q < 8; ++q) {
        tot += sS[q][tid];
        o0 += sO0[q][tid]; o1 += sO1[q][tid]; o2 += sO2[q][tid];
      }
      const float inv = 1.0f / tot;          // tot>0 guaranteed by cr>1
      out[((size_t)b * 3 + 0) * NPIX + i] = o0 * inv;
      out[((size_t)b * 3 + 1) * NPIX + i] = o1 * inv;
      out[((size_t)b * 3 + 2) * NPIX + i] = o2 * inv;
    }
  }
}

// ---------------------------------------------------------------- launch
extern "C" void kernel_launch(void* const* d_in, const int* in_sizes, int n_in,
                              void* d_out, int out_size, void* d_ws, size_t ws_size,
                              hipStream_t stream) {
  const float* gf   = (const float*)d_in[0];   // gray_feature (B,C,H,W)
  const float* rf   = (const float*)d_in[1];   // rgb_feature  (B,C,H,W)
  const float* img  = (const float*)d_in[2];   // rgb_image    (B,3,H,W)
  const float* glab = (const float*)d_in[3];   // gray_label   (B,12,H,W)
  const float* rlab = (const float*)d_in[4];   // rgb_label    (B,12,H,W)
  float* out = (float*)d_out;

  // workspace carve-up (~57 MB): counts, means, f16 fragment arrays
  float* cntg = (float*)d_ws;                          // B*12
  float* cntr = cntg + B_ * NCHN;                      // B*12
  float* mg = cntr + B_ * NCHN;                        // B*12*C
  float* mr = mg + (size_t)B_ * NCHN * C_;             // B*12*C
  _Float16* gnw = (_Float16*)(mr + (size_t)B_ * NCHN * C_);   // 32B-aligned (offset 49344)
  _Float16* rnw = gnw + (size_t)B_ * NCHN * NPIX * C_;

  init_kernel<<<(B_ * 3 * NPIX + 255) / 256, 256, 0, stream>>>(out, B_ * 3 * NPIX);
  stats_kernel<<<dim3(NCHN - 1, B_), 256, 0, stream>>>(gf, rf, glab, rlab, cntg, cntr, mg, mr);
  normalize_kernel<<<dim3(NPIX / 32, NCHN - 1, B_), 256, 0, stream>>>(gf, rf, glab, rlab, mg, mr, gnw, rnw);
  corr_softmax_kernel<<<dim3(NPIX / 16, NCHN - 1, B_), 256, 0, stream>>>(img, glab, rlab, cntg, cntr, gnw, rnw, out);
}
